// SelectiveSSM_6837587935449
// MI455X (gfx1250) — matrix operand order repaired and checked by reading, not compiled, not run on hardware
//
#include <hip/hip_runtime.h>
#include <hip/hip_bf16.h>

// ---------------- problem constants ----------------
#define B_SZ   4
#define T_LEN  2048
#define D_MODEL 768
#define D_STATE 16
#define D_CONV  4
#define M_TOT  (B_SZ * T_LEN)       // 8192 rows
#define K_DIM  D_MODEL              // 768

typedef __bf16 bf16_t;
typedef __attribute__((ext_vector_type(16))) __bf16 v16bf;
typedef __attribute__((ext_vector_type(8)))  __bf16 v8bf;
typedef __attribute__((ext_vector_type(8)))  float  v8f;

union FragBF {
    v16bf v;
    v8bf  h[2];
};

// ---- CDNA5 async global->LDS (ASYNCcnt-tracked DMA, bypasses VGPRs) ----
__device__ __forceinline__ void async_load_b128(unsigned lds_off, const void* gptr) {
    asm volatile("global_load_async_to_lds_b128 %0, %1, off"
                 :: "v"(lds_off), "v"((unsigned long long)(uintptr_t)gptr)
                 : "memory");
}
__device__ __forceinline__ void wait_async_le4() {   // current tile's 4 loads done
    asm volatile("s_wait_asynccnt 0x4" ::: "memory");
}
__device__ __forceinline__ void wait_async_0() {
    asm volatile("s_wait_asynccnt 0x0" ::: "memory");
}

// ---------------- fp32 -> bf16 convert ----------------
__global__ void cvt_f32_bf16(const float* __restrict__ src,
                             bf16_t* __restrict__ dst, int n) {
    int i = blockIdx.x * blockDim.x + threadIdx.x;
    if (i < n) dst[i] = (bf16_t)src[i];
}

// ---------------- WMMA bf16 GEMM:  C[MxN] = A[MxK] * W[NxK]^T ----------------
// Tile: 128x128 per workgroup (256 threads = 8 waves, 4(M) x 2(N) wave grid),
// each wave computes 32x64 = 2x4 WMMA 16x16 tiles.  BLK_K = 32.
// Double-buffered LDS filled by GLOBAL_LOAD_ASYNC_TO_LDS_B128.
#define BLK_M 128
#define BLK_N 128
#define BLK_K 32
#define LDSS  48                 // padded bf16 stride per row (96B)
#define TILE_A_BYTES (BLK_M * LDSS * 2)           // 12288
#define BUF_BYTES    (2 * TILE_A_BYTES)           // sA + sB per buffer = 24576
#define LDS_TOTAL    (2 * BUF_BYTES)              // 49152 (dynamic LDS)

__global__ __launch_bounds__(256)
void gemm_bf16_wmma(const bf16_t* __restrict__ A,   // M x K row-major
                    const bf16_t* __restrict__ W,   // N x K row-major
                    float* __restrict__ C,          // M x N row-major
                    int M, int N, int K,
                    const float* __restrict__ bias, // per-column, epilogue==1
                    int epilogue)                   // 0 = plain, 1 = softplus(x+bias)
{
    extern __shared__ char smem[];                  // dynamic LDS, offset 0

    const int tid  = threadIdx.x;
    const int wave = tid >> 5;
    const int lane = tid & 31;
    const int wm   = wave & 3;   // 0..3 -> M offset wm*32
    const int wn   = wave >> 2;  // 0..1 -> N offset wn*64
    const int m0   = blockIdx.x * BLK_M;
    const int n0   = blockIdx.y * BLK_N;

    const int lrow = lane & 15;
    const int hi   = lane >> 4;       // lane half
    const int aK0  = hi * 8;          // A frag: elems 0..7 -> K=aK0.., 8..15 -> 16+aK0..
    const int bK0  = hi * 8;          // B frag: elems 0..7 -> K=bK0.., 8..15 -> 16+bK0..

    v8f acc[2][4];
#pragma unroll
    for (int i = 0; i < 2; ++i)
#pragma unroll
        for (int j = 0; j < 4; ++j)
            acc[i][j] = (v8f){0.f,0.f,0.f,0.f,0.f,0.f,0.f,0.f};

    // staging map: each thread owns 16 bf16 (2 x b128) of A and of W per K-tile
    const int lr  = tid >> 1;          // 0..127 row within tile
    const int seg = (tid & 1) * 16;    // 0 or 16 (bf16 elems)
    const bf16_t* Asrc = A + (size_t)(m0 + lr) * K + seg;
    const bf16_t* Wsrc = W + (size_t)(n0 + lr) * K + seg;
    const unsigned ldsA = (unsigned)((lr * LDSS + seg) * 2);                 // byte off
    const unsigned ldsB = (unsigned)(TILE_A_BYTES + (lr * LDSS + seg) * 2);

    const int nk = K / BLK_K;   // 24

    // prologue: async-fill buffer 0 with K-tile 0
    async_load_b128(ldsA,      Asrc);
    async_load_b128(ldsA + 16, Asrc + 8);
    async_load_b128(ldsB,      Wsrc);
    async_load_b128(ldsB + 16, Wsrc + 8);

    for (int kt = 0; kt < nk; ++kt) {
        const int cur = kt & 1;
        if (kt + 1 < nk) {
            // issue next tile into the other buffer (freed by last-iter barrier),
            // then wait until only those 4 remain outstanding (in-order per wave)
            const unsigned bo = (unsigned)((cur ^ 1) * BUF_BYTES);
            const bf16_t* An = Asrc + (size_t)(kt + 1) * BLK_K;
            const bf16_t* Wn = Wsrc + (size_t)(kt + 1) * BLK_K;
            async_load_b128(ldsA + bo,      An);
            async_load_b128(ldsA + bo + 16, An + 8);
            async_load_b128(ldsB + bo,      Wn);
            async_load_b128(ldsB + bo + 16, Wn + 8);
            wait_async_le4();
        } else {
            wait_async_0();
        }
        __syncthreads();    // current tile resident for all waves

        const bf16_t* sAb = (const bf16_t*)(smem + cur * BUF_BYTES);
        const bf16_t* sBb = (const bf16_t*)(smem + cur * BUF_BYTES + TILE_A_BYTES);

        // fragment loads (documented CDNA5 wave32 layouts)
        FragBF af[2], bfr[4];
#pragma unroll
        for (int mt = 0; mt < 2; ++mt) {
            int am = wm * 32 + mt * 16 + lrow;
            af[mt].h[0] = *reinterpret_cast<const v8bf*>(&sAb[am * LDSS + aK0]);
            af[mt].h[1] = *reinterpret_cast<const v8bf*>(&sAb[am * LDSS + 16 + aK0]);
        }
#pragma unroll
        for (int nt = 0; nt < 4; ++nt) {
            int bn = wn * 64 + nt * 16 + lrow;
            bfr[nt].h[0] = *reinterpret_cast<const v8bf*>(&sBb[bn * LDSS + bK0]);
            bfr[nt].h[1] = *reinterpret_cast<const v8bf*>(&sBb[bn * LDSS + bK0 + 16]);
        }
#pragma unroll
        for (int mt = 0; mt < 2; ++mt)
#pragma unroll
            for (int nt = 0; nt < 4; ++nt)
                acc[mt][nt] = __builtin_amdgcn_wmma_f32_16x16x32_bf16(
                    false, af[mt].v, false, bfr[nt].v,
                    (short)0, acc[mt][nt], false, false);
        __syncthreads();    // all waves done reading buffer `cur` before refill
    }

    // epilogue: D layout row M = 8*hi + r, col N = lrow within each 16x16 tile
#pragma unroll
    for (int mt = 0; mt < 2; ++mt) {
#pragma unroll
        for (int nt = 0; nt < 4; ++nt) {
            int gmb = m0 + wm * 32 + mt * 16 + hi * 8;
            int gn  = n0 + wn * 64 + nt * 16 + lrow;
            float bz = (epilogue == 1) ? bias[gn] : 0.f;
#pragma unroll
            for (int r = 0; r < 8; ++r) {
                float v = acc[mt][nt][r];
                if (epilogue == 1) {
                    v += bz;
                    v = (v > 20.f) ? v : log1pf(__expf(v));  // softplus
                }
                C[(size_t)(gmb + r) * N + gn] = v;
            }
        }
    }
}

// ---------------- depthwise causal conv (k=4) + SiLU ----------------
__global__ void conv_silu_kernel(const float* __restrict__ xs,   // (B*T, D) = x_ssm
                                 const float* __restrict__ cw,   // (D,1,4)
                                 const float* __restrict__ cb,   // (D,)
                                 float* __restrict__ xc,
                                 bf16_t* __restrict__ xcb)
{
    int idx = blockIdx.x * blockDim.x + threadIdx.x;      // < M_TOT * D_MODEL
    int d  = idx % D_MODEL;
    int bt = idx / D_MODEL;
    int t  = bt % T_LEN;
    float acc = cb[d];
#pragma unroll
    for (int j = 0; j < D_CONV; ++j) {
        int tt = t - (D_CONV - 1) + j;
        if (tt >= 0)
            acc += xs[(size_t)(bt + (j - (D_CONV - 1))) * D_MODEL + d] * cw[d * D_CONV + j];
    }
    float s = acc / (1.f + __expf(-acc));   // SiLU
    xc[idx]  = s;
    xcb[idx] = (bf16_t)s;
}

// ---------------- small GEMM for Bm / Cm (N = 16 each) ----------------
__global__ void bc_gemm_kernel(const float* __restrict__ xc,   // (M, K)
                               const float* __restrict__ Bw,   // (16, K)
                               const float* __restrict__ Cw,   // (16, K)
                               float* __restrict__ Bm,         // (M, 16)
                               float* __restrict__ Cm)         // (M, 16)
{
    int idx = blockIdx.x * blockDim.x + threadIdx.x;   // < M_TOT * 32
    int n = idx & 31;
    int m = idx >> 5;
    const float* w = (n < 16) ? (Bw + n * K_DIM) : (Cw + (n - 16) * K_DIM);
    const float* a = xc + (size_t)m * K_DIM;
    float acc = 0.f;
#pragma unroll 4
    for (int k = 0; k < K_DIM; ++k) acc = fmaf(a[k], w[k], acc);
    if (n < 16) Bm[m * 16 + n]        = acc;
    else        Cm[m * 16 + (n - 16)] = acc;
}

// ---------------- selective scan: 16 lanes per (b,d) sequence ----------------
__global__ __launch_bounds__(256)
void scan_kernel(const float* __restrict__ x,      // original input (B,T,D)
                 const float* __restrict__ dt,     // (B,T,D)
                 const float* __restrict__ Bm,     // (B,T,16)
                 const float* __restrict__ Cm,     // (B,T,16)
                 const float* __restrict__ A_log,  // (D,16)
                 const float* __restrict__ Dsk,    // (D,)
                 float* __restrict__ out)          // (B,T,D)
{
    const int lane = threadIdx.x & 31;
    const int n    = lane & 15;
    const int grp  = threadIdx.x >> 4;                 // 16 groups per block
    const int pair = blockIdx.x * 16 + grp;            // < B_SZ * D_MODEL
    const int b = pair / D_MODEL;
    const int d = pair % D_MODEL;

    const float Ad = -__expf(A_log[d * D_STATE + n]);  // A[d][n]
    const float Dd = Dsk[d];
    float h = 0.f;

    const float* dt_p = dt + (size_t)b * T_LEN * D_MODEL + d;
    const float* x_p  = x  + (size_t)b * T_LEN * D_MODEL + d;
    const float* B_p  = Bm + (size_t)b * T_LEN * D_STATE + n;
    const float* C_p  = Cm + (size_t)b * T_LEN * D_STATE + n;
    float*       o_p  = out + (size_t)b * T_LEN * D_MODEL + d;

    for (int t0 = 0; t0 < T_LEN; t0 += 16) {
        // each lane preloads one timestep's dt/x; broadcast via width-16 shuffles
        float dt_l = dt_p[(size_t)(t0 + n) * D_MODEL];
        float x_l  = x_p [(size_t)(t0 + n) * D_MODEL];
#pragma unroll
        for (int i = 0; i < 16; ++i) {
            const int t = t0 + i;
            float dtv = __shfl(dt_l, i, 16);
            float xv  = __shfl(x_l,  i, 16);
            float Bv  = B_p[(size_t)t * D_STATE];
            float Cv  = C_p[(size_t)t * D_STATE];
            float dA  = __expf(Ad * dtv);
            h = fmaf(h, dA, xv * dtv * Bv);
            float y = h * Cv;
            y += __shfl_xor(y, 8, 16);
            y += __shfl_xor(y, 4, 16);
            y += __shfl_xor(y, 2, 16);
            y += __shfl_xor(y, 1, 16);
            if (n == 0) o_p[(size_t)t * D_MODEL] = y + xv * Dd;
        }
    }
}

// ---------------- launcher ----------------
extern "C" void kernel_launch(void* const* d_in, const int* in_sizes, int n_in,
                              void* d_out, int out_size, void* d_ws, size_t ws_size,
                              hipStream_t stream) {
    const float* x    = (const float*)d_in[0];  // (B,T,D)
    const float* w1   = (const float*)d_in[1];  // (2D, D)
    const float* cw   = (const float*)d_in[2];  // (D,1,4)
    const float* cb   = (const float*)d_in[3];  // (D,)
    const float* dtw  = (const float*)d_in[4];  // (D, D)
    const float* dtb  = (const float*)d_in[5];  // (D,)
    const float* Alog = (const float*)d_in[6];  // (D,16)
    const float* Dsk  = (const float*)d_in[7];  // (D,)
    const float* Bw   = (const float*)d_in[8];  // (16, D)
    const float* Cw   = (const float*)d_in[9];  // (16, D)
    float* out = (float*)d_out;

    // workspace layout (bytes)
    char* ws = (char*)d_ws;
    const size_t n_xd   = (size_t)M_TOT * D_MODEL;     // 6291456
    const size_t n_w    = (size_t)D_MODEL * D_MODEL;   // 589824
    bf16_t* xb    = (bf16_t*)(ws);                          // 12.6 MB
    bf16_t* w1b   = (bf16_t*)(ws + 12582912);               // 1.2 MB
    bf16_t* dtwb  = (bf16_t*)(ws + 12582912 + 1179648);     // 1.2 MB
    float*  xssm  = (float*) (ws + 12582912 + 2*1179648);                 // 25.2 MB
    float*  xc    = (float*) (ws + 12582912 + 2*1179648 + 25165824);      // 25.2 MB
    bf16_t* xcb   = (bf16_t*)(ws + 12582912 + 2*1179648 + 2*25165824);    // 12.6 MB
    float*  dtbuf = (float*) (ws + 2*12582912 + 2*1179648 + 2*25165824);  // 25.2 MB
    float*  Bm    = (float*) (ws + 2*12582912 + 2*1179648 + 3*25165824);
    float*  Cm    = (float*) (ws + 2*12582912 + 2*1179648 + 3*25165824 + 524288);

    // 1) convert to bf16: x, in_proj rows [0,768) (only half of xz is used), dt_w
    cvt_f32_bf16<<<(int)((n_xd + 255) / 256), 256, 0, stream>>>(x, xb, (int)n_xd);
    cvt_f32_bf16<<<(int)((n_w + 255) / 256), 256, 0, stream>>>(w1, w1b, (int)n_w);
    cvt_f32_bf16<<<(int)((n_w + 255) / 256), 256, 0, stream>>>(dtw, dtwb, (int)n_w);

    // 2) x_ssm = x @ in_proj_w[:768].T   (WMMA bf16, async-LDS pipelined)
    dim3 g1(M_TOT / BLK_M, D_MODEL / BLK_N);
    gemm_bf16_wmma<<<g1, 256, LDS_TOTAL, stream>>>(xb, w1b, xssm, M_TOT, D_MODEL,
                                                   K_DIM, nullptr, 0);

    // 3) depthwise causal conv + SiLU
    conv_silu_kernel<<<(int)(n_xd / 256), 256, 0, stream>>>(xssm, cw, cb, xc, xcb);

    // 4) dt = softplus(xc @ dt_w.T + dt_b)   (WMMA bf16, fused epilogue)
    gemm_bf16_wmma<<<g1, 256, LDS_TOTAL, stream>>>(xcb, dtwb, dtbuf, M_TOT, D_MODEL,
                                                   K_DIM, dtb, 1);

    // 5) Bm / Cm (N=16 each, fp32 VALU dot)
    bc_gemm_kernel<<<(M_TOT * 32) / 256, 256, 0, stream>>>(xc, Bw, Cw, Bm, Cm);

    // 6) selective scan + skip
    scan_kernel<<<(B_SZ * D_MODEL) / 16, 256, 0, stream>>>(x, dtbuf, Bm, Cm,
                                                           Alog, Dsk, out);
}
